// MambaLayer_35364760715771
// MI455X (gfx1250) — matrix operand order repaired and checked by reading, not compiled, not run on hardware
//
#include <hip/hip_runtime.h>
#include <hip/hip_bf16.h>

// ---------------------------------------------------------------------------
// Problem constants (from reference)
// ---------------------------------------------------------------------------
#define BSZ     2
#define CDIM    384           // DIM
#define LSEQ    2048          // 8*16*16
#define DI      768           // D_INNER
#define TWO_DI  1536
#define NSTATE  16            // D_STATE
#define DTRANK  24
#define MROWS   (BSZ * LSEQ)  // 4096 GEMM rows

typedef __attribute__((ext_vector_type(16))) _Float16 v16h;
typedef __attribute__((ext_vector_type(8)))  _Float16 v8h;
typedef __attribute__((ext_vector_type(8)))  float    v8f;
typedef __attribute__((ext_vector_type(4)))  unsigned v4u;
typedef __attribute__((ext_vector_type(4)))  int      v4i;
typedef __attribute__((ext_vector_type(8)))  int      v8i;

// branch sequence index map: t (branch order) -> l (canonical order)
__device__ __forceinline__ int seqmap(int mode, int t) {
    if (mode == 0) return t;                       // forward
    if (mode == 1) return (LSEQ - 1) - t;          // backward (flip)
    return ((t & 7) << 8) + (t >> 3);              // slice: l = (t%8)*256 + t/8
}

// ---------------------------------------------------------------------------
// Tensor Data Mover: async 2D f32 tile load Global -> LDS.
// D# built per CDNA5 ISA 8.3/8.4: group0 = {count|lds_addr, global_addr|type=2},
// group1 = {data_size=4B, tensor_dim0/1, tile_dim0/1, tensor_dim0_stride}.
// Issued by one wave; completion via TENSORcnt.
// ---------------------------------------------------------------------------
__device__ __forceinline__ void tdm_load_2d_f32(unsigned ldsOff, const void* gptr,
                                                unsigned tensor_d0, unsigned tensor_d1,
                                                unsigned tile_d0, unsigned tile_d1,
                                                unsigned long long stride0) {
    union { v4u v; unsigned long long q[2]; } g0;
    union { v8i v; unsigned long long q[4]; } g1;
    union { v4i v; unsigned long long q[2]; } gz4;
    unsigned long long ga = (unsigned long long)(size_t)gptr;
    g0.q[0] = 1ull | ((unsigned long long)ldsOff << 32);
    g0.q[1] = (ga & 0x01FFFFFFFFFFFFFFull) | (2ull << 62);
    g1.q[0] = (2ull << 16) | ((unsigned long long)(tensor_d0 & 0xFFFFu) << 48);
    g1.q[1] = ((unsigned long long)(tensor_d0 >> 16) & 0xFFFFull)
            | ((unsigned long long)tensor_d1 << 16)
            | ((unsigned long long)(tile_d0 & 0xFFFFu) << 48);
    g1.q[2] = (unsigned long long)(tile_d1 & 0xFFFFu)
            | ((stride0 & 0xFFFFFFFFull) << 32);
    g1.q[3] = (stride0 >> 32) & 0xFFFFull;
    gz4.q[0] = 0; gz4.q[1] = 0;
#if __has_include(<hip/amd_detail/amd_gfx1250_TDM.h>)
    union { v8i v; unsigned long long q[4]; } gz8;
    gz8.q[0] = 0; gz8.q[1] = 0; gz8.q[2] = 0; gz8.q[3] = 0;
    __builtin_amdgcn_tensor_load_to_lds(g0.v, g1.v, gz4.v, gz4.v, gz8.v, 0);
#else
    __builtin_amdgcn_tensor_load_to_lds(g0.v, g1.v, gz4.v, gz4.v, 0);
#endif
}

// ---------------------------------------------------------------------------
// WMMA 16x64 tile core: 4 accumulators share one A fragment per K-step.
// Packing: contiguous k = 16*(lane/16) + s in both A and B (16-bit A 16x32 / B 32x16).
// ---------------------------------------------------------------------------
template<int K, int LDA, int LDB>
__device__ __forceinline__ void wmma_tile16x64(const _Float16* __restrict__ A,
                                               const _Float16* __restrict__ Wn,
                                               v8f acc[4]) {
    const int lane = threadIdx.x & 31;
    const int hi   = lane >> 4;
    const int lx   = lane & 15;
#pragma unroll
    for (int k0 = 0; k0 < K; k0 += 32) {
        const _Float16* ap = A + lx * LDA + k0 + hi * 16;
        v8h alo = *(const v8h*)(ap);
        v8h ahi = *(const v8h*)(ap + 8);
        v16h a;
#pragma unroll
        for (int i = 0; i < 8; ++i) { a[i] = alo[i]; a[i + 8] = ahi[i]; }
#pragma unroll
        for (int jt = 0; jt < 4; ++jt) {
            v16h b = *(const v16h*)(Wn + (jt * 16 + lx) * LDB + k0 + hi * 16);
            acc[jt] = __builtin_amdgcn_wmma_f32_16x16x32_f16(
                          false, a, false, b, (short)0, acc[jt], false, false);
        }
    }
}

// ---------------------------------------------------------------------------
// Helper conversion / padding kernels
// ---------------------------------------------------------------------------
__global__ void k_f32_to_f16(const float* __restrict__ src, _Float16* __restrict__ dst, int n) {
    int i = blockIdx.x * blockDim.x + threadIdx.x;
    if (i < n) dst[i] = (_Float16)src[i];
}

__global__ void k_pad_rows(const float* __restrict__ src, _Float16* __restrict__ dst,
                           int srcRows, int dstRows, int cols) {
    int i = blockIdx.x * blockDim.x + threadIdx.x;
    if (i >= dstRows * cols) return;
    int r = i / cols;
    dst[i] = (r < srcRows) ? (_Float16)src[i] : (_Float16)0.f;
}

__global__ void k_pad_cols(const float* __restrict__ src, _Float16* __restrict__ dst,
                           int rows, int srcCols, int dstCols) {
    int i = blockIdx.x * blockDim.x + threadIdx.x;
    if (i >= rows * dstCols) return;
    int r = i / dstCols, c = i % dstCols;
    dst[i] = (c < srcCols) ? (_Float16)src[r * srcCols + c] : (_Float16)0.f;
}

// combine up to 3 branch outputs and convert to f16 for out_proj
__global__ void k_combine_f16(const float* __restrict__ y0, const float* __restrict__ y1,
                              const float* __restrict__ y2, int nb,
                              _Float16* __restrict__ dst, int n) {
    int i = blockIdx.x * blockDim.x + threadIdx.x;
    if (i >= n) return;
    float s = y0[i];
    if (nb > 1) s += y1[i] + y2[i];
    dst[i] = (_Float16)s;
}

// ---------------------------------------------------------------------------
// LayerNorm over channels: x (B,C,L) -> xn_h (M x C, f16), one wave per row
// ---------------------------------------------------------------------------
__global__ void k_layernorm(const float* __restrict__ x,
                            const float* __restrict__ lnw,
                            const float* __restrict__ lnb,
                            _Float16* __restrict__ xnh) {
    int row  = blockIdx.x * (blockDim.x >> 5) + (threadIdx.x >> 5);
    int lane = threadIdx.x & 31;
    int b = row >> 11, l = row & (LSEQ - 1);
    const float* xp = x + (size_t)b * CDIM * LSEQ + l;
    float v[CDIM / 32];
    float s = 0.f, s2 = 0.f;
#pragma unroll
    for (int i = 0; i < CDIM / 32; ++i) {
        int c = lane + i * 32;
        float t = xp[(size_t)c * LSEQ];
        v[i] = t; s += t; s2 += t * t;
    }
#pragma unroll
    for (int off = 16; off; off >>= 1) {
        s  += __shfl_xor(s,  off, 32);
        s2 += __shfl_xor(s2, off, 32);
    }
    float mu   = s * (1.f / CDIM);
    float var  = s2 * (1.f / CDIM) - mu * mu;
    float rstd = rsqrtf(var + 1e-5f);
    _Float16* op = xnh + (size_t)row * CDIM;
#pragma unroll
    for (int i = 0; i < CDIM / 32; ++i) {
        int c = lane + i * 32;
        op[c] = (_Float16)(((v[i] - mu) * rstd) * lnw[c] + lnb[c]);
    }
}

// ---------------------------------------------------------------------------
// in_proj GEMM: xz[b][e][l]  (M=4096, N=1536, K=384)
// ---------------------------------------------------------------------------
__global__ void k_inproj(const _Float16* __restrict__ xnh,
                         const _Float16* __restrict__ wh,
                         float* __restrict__ xz) {
    const int NG = TWO_DI / 64;                               // 24
    int g  = blockIdx.x * (blockDim.x >> 5) + (threadIdx.x >> 5);
    int mt = g / NG, ng = g % NG;
    v8f acc[4] = {};
    wmma_tile16x64<CDIM, CDIM, CDIM>(xnh + (size_t)mt * 16 * CDIM,
                                     wh  + (size_t)ng * 64 * CDIM, acc);
    int lane = threadIdx.x & 31, hi = lane >> 4, lx = lane & 15;
#pragma unroll
    for (int jt = 0; jt < 4; ++jt) {
        int n = ng * 64 + jt * 16 + lx;
#pragma unroll
        for (int j = 0; j < 8; ++j) {
            int m = mt * 16 + j + 8 * hi;
            int b = m >> 11, l = m & (LSEQ - 1);
            xz[((size_t)b * TWO_DI + n) * LSEQ + l] = acc[jt][j];
        }
    }
}

// ---------------------------------------------------------------------------
// Depthwise causal conv (k=4) + SiLU in branch order, plus precomputed
// z-gate g = silu(z); all outputs coalesced [b][t][d]
// ---------------------------------------------------------------------------
__global__ void k_conv_silu(const float* __restrict__ xz,
                            const float* __restrict__ cw,
                            const float* __restrict__ cb,
                            float* __restrict__ xcT,
                            _Float16* __restrict__ xcTh,
                            float* __restrict__ gT,
                            int mode) {
    int bd = blockIdx.x;
    int b = bd / DI, d = bd % DI;
    const float* xs = xz + ((size_t)b * TWO_DI + d) * LSEQ;        // x half
    const float* zs = xz + ((size_t)b * TWO_DI + DI + d) * LSEQ;   // z half
    float w0 = cw[d * 4 + 0], w1 = cw[d * 4 + 1], w2 = cw[d * 4 + 2], w3 = cw[d * 4 + 3];
    float bias = cb[d];
    for (int t = threadIdx.x; t < LSEQ; t += blockDim.x) {
        float acc = bias + w3 * xs[seqmap(mode, t)];
        if (t >= 1) acc += w2 * xs[seqmap(mode, t - 1)];
        if (t >= 2) acc += w1 * xs[seqmap(mode, t - 2)];
        if (t >= 3) acc += w0 * xs[seqmap(mode, t - 3)];
        float sv = acc / (1.f + __expf(-acc));                     // SiLU
        float z  = zs[seqmap(mode, t)];
        float g  = z / (1.f + __expf(-z));                         // silu(z)
        size_t idx = ((size_t)b * LSEQ + t) * DI + d;
        xcT[idx]  = sv;
        xcTh[idx] = (_Float16)sv;
        gT[idx]   = g;
    }
}

// ---------------------------------------------------------------------------
// xproj GEMM: dbl[m][e] (M=4096, N=56->64, K=768); emits f16 dt rows (K->32)
// ---------------------------------------------------------------------------
__global__ void k_xproj(const _Float16* __restrict__ xcTh,
                        const _Float16* __restrict__ xwh,   // 64 x 768 padded
                        float* __restrict__ dbl,            // M x 56
                        _Float16* __restrict__ dth) {       // M x 32 (24 real)
    int mt = blockIdx.x * (blockDim.x >> 5) + (threadIdx.x >> 5);
    v8f acc[4] = {};
    wmma_tile16x64<DI, DI, DI>(xcTh + (size_t)mt * 16 * DI, xwh, acc);
    int lane = threadIdx.x & 31, hi = lane >> 4, lx = lane & 15;
#pragma unroll
    for (int jt = 0; jt < 4; ++jt) {
        int n = jt * 16 + lx;
#pragma unroll
        for (int j = 0; j < 8; ++j) {
            int m = mt * 16 + j + 8 * hi;
            float v = acc[jt][j];
            if (n < DTRANK + 2 * NSTATE) dbl[(size_t)m * 56 + n] = v;
            if (n < 32) dth[(size_t)m * 32 + n] = (n < DTRANK) ? (_Float16)v : (_Float16)0.f;
        }
    }
}

// ---------------------------------------------------------------------------
// dtproj GEMM + bias + softplus: delta[m][d]  (M=4096, N=768, K=24->32)
// ---------------------------------------------------------------------------
__global__ void k_dtproj(const _Float16* __restrict__ dth,
                         const _Float16* __restrict__ dwh,  // 768 x 32 padded
                         const float* __restrict__ dbias,
                         float* __restrict__ delta) {
    const int NG = DI / 64;                                  // 12
    int g  = blockIdx.x * (blockDim.x >> 5) + (threadIdx.x >> 5);
    int mt = g / NG, ng = g % NG;
    v8f acc[4] = {};
    wmma_tile16x64<32, 32, 32>(dth + (size_t)mt * 16 * 32,
                               dwh + (size_t)ng * 64 * 32, acc);
    int lane = threadIdx.x & 31, hi = lane >> 4, lx = lane & 15;
#pragma unroll
    for (int jt = 0; jt < 4; ++jt) {
        int n = ng * 64 + jt * 16 + lx;
        float bn = dbias[n];
#pragma unroll
        for (int j = 0; j < 8; ++j) {
            int m = mt * 16 + j + 8 * hi;
            float v = acc[jt][j] + bn;
            float sp = (v > 20.f) ? v : log1pf(__expf(v));
            delta[(size_t)m * DI + n] = sp;
        }
    }
}

// ---------------------------------------------------------------------------
// Selective scan, optionally fused over branches (grid = nb*6 blocks).
// One lane per (b,d) chain, 16 states in VGPRs. B/C staged to LDS via a
// double-buffered TDM pipeline: issue chunk n+1, s_wait_tensorcnt(1) for
// chunk n (TDM ops from one wave complete in order).
// ---------------------------------------------------------------------------
__global__ void k_scan(const float* __restrict__ xcT0,
                       const float* __restrict__ delta0,
                       const float* __restrict__ gT0,
                       const float* __restrict__ dbl0,
                       const float* __restrict__ Al0, const float* __restrict__ Al1,
                       const float* __restrict__ Al2,
                       const float* __restrict__ Dv0, const float* __restrict__ Dv1,
                       const float* __restrict__ Dv2,
                       float* __restrict__ y0,
                       size_t strideBig, size_t strideDbl,
                       int modeBase, int accumulate) {
    __shared__ float sBC[2][32 * 32];               // double-buffered 32x(B16|C16)
    const int BPB = BSZ * (DI / 256);               // 6 blocks per branch
    int brl  = blockIdx.x / BPB;
    int sub  = blockIdx.x % BPB;
    int mode = modeBase + brl;
    int b = sub / (DI / 256);
    int d = (sub % (DI / 256)) * 256 + threadIdx.x;
    const float* xcT   = xcT0   + strideBig * brl;
    const float* delta = delta0 + strideBig * brl;
    const float* gT    = gT0    + strideBig * brl;
    const float* dbl   = dbl0   + strideDbl * brl;
    float*       yb    = y0     + strideBig * brl;
    const float* A_log = (brl == 0) ? Al0 : (brl == 1) ? Al1 : Al2;
    const float* Dvec  = (brl == 0) ? Dv0 : (brl == 1) ? Dv1 : Dv2;

    float An[NSTATE], h[NSTATE];
#pragma unroll
    for (int n = 0; n < NSTATE; ++n) {
        An[n] = -__expf(A_log[d * NSTATE + n]);
        h[n]  = 0.f;
    }
    float Dp = Dvec[d];

    // prologue: DMA chunk 0 into buffer 0
    if (threadIdx.x < 32)
        tdm_load_2d_f32((unsigned)(size_t)&sBC[0][0],
                        &dbl[((size_t)b * LSEQ) * 56 + DTRANK],
                        32, 32, 32, 32, 56);

    for (int t0 = 0; t0 < LSEQ; t0 += 32) {
        int cur = (t0 >> 5) & 1;
        __syncthreads();                   // all readers done with buf cur^1
        if (threadIdx.x < 32) {
            if (t0 + 32 < LSEQ) {          // DMA next chunk into other buffer
                tdm_load_2d_f32((unsigned)(size_t)&sBC[cur ^ 1][0],
                                &dbl[((size_t)b * LSEQ + t0 + 32) * 56 + DTRANK],
                                32, 32, 32, 32, 56);
                __builtin_amdgcn_s_wait_tensorcnt(1);   // chunk t0 resident
            } else {
                __builtin_amdgcn_s_wait_tensorcnt(0);
            }
        }
        __syncthreads();
        if (t0 + 32 < LSEQ)   // speculative prefetch of next delta rows
            __builtin_prefetch(&delta[((size_t)b * LSEQ + t0 + 32) * DI + d], 0, 1);
#pragma unroll 4
        for (int tt = 0; tt < 32; ++tt) {
            int t = t0 + tt;
            size_t ridx = ((size_t)b * LSEQ + t) * DI + d;
            float dlt = delta[ridx];
            float u   = xcT[ridx];
            float g   = gT[ridx];
            float du  = dlt * u;
            const float* bc = &sBC[cur][tt * 32];
            float y = 0.f;
#pragma unroll
            for (int n = 0; n < NSTATE; ++n) {
                h[n] = __expf(dlt * An[n]) * h[n] + du * bc[n];
                y += h[n] * bc[16 + n];
            }
            int l = seqmap(mode, t);
            float out = (y + u * Dp) * g;
            size_t oidx = ((size_t)b * LSEQ + l) * DI + d;
            if (accumulate) yb[oidx] += out; else yb[oidx] = out;
        }
    }
}

// ---------------------------------------------------------------------------
// out_proj GEMM: out[b][o][l]  (M=4096, N=384, K=768)
// ---------------------------------------------------------------------------
__global__ void k_outproj(const _Float16* __restrict__ yh,
                          const _Float16* __restrict__ wh,
                          float* __restrict__ out) {
    const int NG = CDIM / 64;                                // 6
    int g  = blockIdx.x * (blockDim.x >> 5) + (threadIdx.x >> 5);
    int mt = g / NG, ng = g % NG;
    v8f acc[4] = {};
    wmma_tile16x64<DI, DI, DI>(yh + (size_t)mt * 16 * DI,
                               wh + (size_t)ng * 64 * DI, acc);
    int lane = threadIdx.x & 31, hi = lane >> 4, lx = lane & 15;
#pragma unroll
    for (int jt = 0; jt < 4; ++jt) {
        int n = ng * 64 + jt * 16 + lx;
#pragma unroll
        for (int j = 0; j < 8; ++j) {
            int m = mt * 16 + j + 8 * hi;
            int b = m >> 11, l = m & (LSEQ - 1);
            out[((size_t)b * CDIM + n) * LSEQ + l] = acc[jt][j];
        }
    }
}

// ---------------------------------------------------------------------------
// Orchestration
// ---------------------------------------------------------------------------
extern "C" void kernel_launch(void* const* d_in, const int* in_sizes, int n_in,
                              void* d_out, int out_size, void* d_ws, size_t ws_size,
                              hipStream_t stream) {
    const float* x          = (const float*)d_in[0];
    const float* ln_w       = (const float*)d_in[1];
    const float* ln_b       = (const float*)d_in[2];
    const float* in_proj_w  = (const float*)d_in[3];
    const float* out_proj_w = (const float*)d_in[4];
    const float *cw[3], *cb[3], *xw[3], *dw[3], *db[3], *Al[3], *Dd[3];
    for (int m = 0; m < 3; ++m) {
        int base = 5 + 7 * m;                 // f @5, b @12, s @19
        cw[m] = (const float*)d_in[base + 0];
        cb[m] = (const float*)d_in[base + 1];
        xw[m] = (const float*)d_in[base + 2];
        dw[m] = (const float*)d_in[base + 3];
        db[m] = (const float*)d_in[base + 4];
        Al[m] = (const float*)d_in[base + 5];
        Dd[m] = (const float*)d_in[base + 6];
    }
    float* out = (float*)d_out;

    // fused-branch path needs per-branch activation buffers (~210 MB);
    // choose deterministically from ws_size.
    const int NB = (ws_size >= (size_t)230 * 1024 * 1024) ? 3 : 1;

    char* p = (char*)d_ws;
    auto alloc = [&](size_t bytes) -> void* {
        void* r = (void*)p;
        p += (bytes + 255) & ~(size_t)255;
        return r;
    };
    _Float16* xnh    = (_Float16*)alloc((size_t)MROWS * CDIM * 2);
    _Float16* w_in_h = (_Float16*)alloc((size_t)TWO_DI * CDIM * 2);
    _Float16* w_out_h= (_Float16*)alloc((size_t)CDIM * DI * 2);
    _Float16* xw_h[3]; _Float16* dw_h[3];
    for (int m = 0; m < 3; ++m) {
        xw_h[m] = (_Float16*)alloc((size_t)64 * DI * 2);
        dw_h[m] = (_Float16*)alloc((size_t)DI * 32 * 2);
    }
    float*    xz = (float*)alloc((size_t)BSZ * TWO_DI * LSEQ * 4);
    _Float16* yh = (_Float16*)alloc((size_t)MROWS * DI * 2);

    const size_t SBIG = (size_t)MROWS * DI;      // per-branch big-buffer elems
    const size_t SDBL = (size_t)MROWS * 56;      // per-branch dbl elems
    // contiguous per-branch arrays so the fused scan can use base + brl*stride
    float*    xcT   = (float*)   alloc(NB * SBIG * 4);
    _Float16* xcTh  = (_Float16*)alloc(NB * SBIG * 2);
    float*    gT    = (float*)   alloc(NB * SBIG * 4);
    float*    dbl   = (float*)   alloc(NB * SDBL * 4);
    _Float16* dth   = (_Float16*)alloc((size_t)NB * MROWS * 32 * 2);
    float*    delta = (float*)   alloc(NB * SBIG * 4);
    float*    ybr   = (float*)   alloc(NB * SBIG * 4);

    // 1) weight conversion / padding
    k_f32_to_f16<<<(TWO_DI * CDIM + 255) / 256, 256, 0, stream>>>(in_proj_w, w_in_h, TWO_DI * CDIM);
    k_f32_to_f16<<<(CDIM * DI + 255) / 256, 256, 0, stream>>>(out_proj_w, w_out_h, CDIM * DI);
    for (int m = 0; m < 3; ++m) {
        k_pad_rows<<<(64 * DI + 255) / 256, 256, 0, stream>>>(xw[m], xw_h[m], 56, 64, DI);
        k_pad_cols<<<(DI * 32 + 255) / 256, 256, 0, stream>>>(dw[m], dw_h[m], DI, DTRANK, 32);
    }

    // 2) LayerNorm
    k_layernorm<<<MROWS / 8, 256, 0, stream>>>(x, ln_w, ln_b, xnh);

    // 3) in_proj WMMA GEMM
    k_inproj<<<(MROWS / 16) * (TWO_DI / 64) / 8, 256, 0, stream>>>(xnh, w_in_h, xz);

    if (NB == 3) {
        // fused path: per-branch GEMM pipeline, then one scan launch (grid 18)
        for (int m = 0; m < 3; ++m) {
            k_conv_silu<<<BSZ * DI, 256, 0, stream>>>(xz, cw[m], cb[m],
                xcT + m * SBIG, xcTh + m * SBIG, gT + m * SBIG, m);
            k_xproj<<<(MROWS / 16) / 8, 256, 0, stream>>>(xcTh + m * SBIG, xw_h[m],
                dbl + m * SDBL, dth + (size_t)m * MROWS * 32);
            k_dtproj<<<(MROWS / 16) * (DI / 64) / 8, 256, 0, stream>>>(
                dth + (size_t)m * MROWS * 32, dw_h[m], db[m], delta + m * SBIG);
        }
        k_scan<<<3 * BSZ * (DI / 256), 256, 0, stream>>>(
            xcT, delta, gT, dbl, Al[0], Al[1], Al[2], Dd[0], Dd[1], Dd[2],
            ybr, SBIG, SDBL, /*modeBase=*/0, /*accumulate=*/0);
        k_combine_f16<<<(MROWS * DI + 255) / 256, 256, 0, stream>>>(
            ybr, ybr + SBIG, ybr + 2 * SBIG, 3, yh, MROWS * DI);
    } else {
        // sequential fallback (shared per-branch buffers, accumulate into ybr)
        for (int m = 0; m < 3; ++m) {
            k_conv_silu<<<BSZ * DI, 256, 0, stream>>>(xz, cw[m], cb[m], xcT, xcTh, gT, m);
            k_xproj<<<(MROWS / 16) / 8, 256, 0, stream>>>(xcTh, xw_h[m], dbl, dth);
            k_dtproj<<<(MROWS / 16) * (DI / 64) / 8, 256, 0, stream>>>(dth, dw_h[m], db[m], delta);
            k_scan<<<BSZ * (DI / 256), 256, 0, stream>>>(
                xcT, delta, gT, dbl, Al[m], Al[m], Al[m], Dd[m], Dd[m], Dd[m],
                ybr, 0, 0, /*modeBase=*/m, /*accumulate=*/(m > 0) ? 1 : 0);
        }
        k_combine_f16<<<(MROWS * DI + 255) / 256, 256, 0, stream>>>(
            ybr, ybr, ybr, 1, yh, MROWS * DI);
    }

    // 5) out_proj WMMA GEMM
    k_outproj<<<(MROWS / 16) * (CDIM / 64) / 8, 256, 0, stream>>>(yh, w_out_h, out);

    (void)in_sizes; (void)n_in; (void)out_size;
}
